// ALIGNNCF_2156073582925
// MI455X (gfx1250) — matrix operand order repaired and checked by reading, not compiled, not run on hardware
//
#include <hip/hip_runtime.h>

typedef __attribute__((ext_vector_type(16))) __bf16 v16bf;
typedef __attribute__((ext_vector_type(8)))  float  v8f;
typedef __bf16 bf16;

// ---------------------------------------------------------------- helpers ---

__device__ __forceinline__ float sp_f(float x) {
  // stable softplus log(1+exp(x))
  return (x > 20.f) ? x : log1pf(__expf(x));
}

// 16-bit A-matrix 16x32 fragment layout (ISA 7.12.2):
// lanes 0-15: VGPR0-3 hold K=0..7 (2 packed each), VGPR4-7 hold K=16..23
// lanes 16-31: K=8..15 / K=24..31
// -> per lane the two 8-element runs are contiguous in a row-major [16 x KPAD]
//    tile, so this lowers to two ds_load_b128 (verified in disasm).
__device__ __forceinline__ int kofs_a(int v, int half) {
  return (v < 4) ? (2 * v + 8 * half) : (16 + 2 * (v - 4) + 8 * half);
}
// contiguous k order, not used here: lanes 0-15 hold K=0..15 across 8 VGPRs (2 packed),
// lanes 16-31 hold K=16..31.
__device__ __forceinline__ int kofs_b(int v, int half) {
  return 16 * half + 2 * v;
}

__device__ __forceinline__ v16bf load_a_frag(const bf16* At, int lda, int kbase, int lane) {
  const int m = lane & 15, half = lane >> 4;
  v16bf a;
#pragma unroll
  for (int v = 0; v < 8; ++v) {
    const int kb = kbase + kofs_a(v, half);
    a[2 * v]     = At[m * lda + kb];
    a[2 * v + 1] = At[m * lda + kb + 1];
  }
  return a;
}

// Weights are staged into LDS pre-swizzled into fragment order:
//   Wl[((kt*4 + t)*32 + lane)*16 + j],  j = 2*v+p,
//   K = kt*32 + kofs_a(v, lane>>4) + p,  n = 16*t + (lane&15)
// so a B fragment is one aligned 32-byte contiguous load (2x ds_load_b128).
template <int KPAD>
__device__ __forceinline__ void stage_w_swz(const float* __restrict__ W, int fin, int fout,
                                            bf16* Wl, int tid) {
  for (int i = tid; i < KPAD * 64; i += 256) {
    const int j    = i & 15;
    const int lane = (i >> 4) & 31;
    const int t    = (i >> 9) & 3;
    const int kt   = i >> 11;
    const int v = j >> 1, p = j & 1, half = lane >> 4;
    const int k = kt * 32 + kofs_a(v, half) + p;
    const int n = 16 * t + (lane & 15);
    Wl[i] = (bf16)((k < fin && n < fout) ? W[k * fout + n] : 0.f);
  }
}

__device__ __forceinline__ v16bf load_b_frag_swz(const bf16* Wl, int kt, int t, int lane) {
  return *(const v16bf*)(Wl + (((kt * 4 + t) * 32 + lane) << 4));
}

// one wave: [16 x KPAD] (bf16, LDS) @ [KPAD x 64] (swizzled bf16, LDS) -> acc[4]
template <int KPAD>
__device__ __forceinline__ void wave_gemm64(const bf16* At, const bf16* Wl, int lane, v8f acc[4]) {
#pragma unroll
  for (int kt = 0; kt < KPAD / 32; ++kt) {
    v16bf a = load_a_frag(At, KPAD, kt * 32, lane);
#pragma unroll
    for (int t = 0; t < 4; ++t) {
      v16bf b = load_b_frag_swz(Wl, kt, t, lane);
      acc[t] = __builtin_amdgcn_wmma_f32_16x16x32_bf16(false, a, false, b, (short)0, acc[t],
                                                       false, false);
    }
  }
}

// ---------------------------------------------------------------- kernels ---

// y = RBF(|r|, 0, 8, 40)   gamma = 39/8
__global__ __launch_bounds__(256) void rbf_y_kernel(const float* __restrict__ r, int E,
                                                    float* __restrict__ y) {
  int e = blockIdx.x * 256 + threadIdx.x;
  if (e < E) {
    float a = r[3 * (size_t)e], b = r[3 * (size_t)e + 1], c = r[3 * (size_t)e + 2];
    float bl = sqrtf(a * a + b * b + c * c);
    const float step = 8.f / 39.f, gamma = 39.f / 8.f;
#pragma unroll 8
    for (int j = 0; j < 40; ++j) {
      float d = bl - j * step;
      y[(size_t)e * 40 + j] = __expf(-gamma * d * d);
    }
  }
}

// generic: dst[nrows, fout] = act(src[nrows, fin] @ W[fin, fout] + bias)
// ACT: 0=none 1=softplus 2=softplus^2 3=relu(bn) with gamma/beta
template <int KPAD, int ACT>
__global__ __launch_bounds__(256) void gemm_rows_kernel(
    const float* __restrict__ src, int nrows, int fin, const float* __restrict__ W,
    const float* __restrict__ bias, int fout, const float* __restrict__ gamma,
    const float* __restrict__ beta, float* __restrict__ dst) {
  __shared__ alignas(32) bf16 Wl[KPAD * 64];
  __shared__ float bl[64], gl[64], bel[64];
  __shared__ alignas(32) bf16 At[8][16 * KPAD];
  const int tid = threadIdx.x, lane = tid & 31, wave = tid >> 5;

  stage_w_swz<KPAD>(W, fin, fout, Wl, tid);
  if (tid < 64) {
    bl[tid] = (tid < fout) ? bias[tid] : 0.f;
    if (ACT == 3) { gl[tid] = gamma[tid]; bel[tid] = beta[tid]; }
  }
  const int r0 = (blockIdx.x * 8 + wave) * 16;
  bf16* at = At[wave];
  for (int i = lane; i < 16 * KPAD; i += 32) {
    int rr = i / KPAD, cc = i - rr * KPAD;
    float v = 0.f;
    if (r0 + rr < nrows && cc < fin) v = src[(size_t)(r0 + rr) * fin + cc];
    at[i] = (bf16)v;
  }
  __syncthreads();

  v8f acc[4] = {};
  wave_gemm64<KPAD>(at, Wl, lane, acc);

  const int colb = lane & 15, half = lane >> 4;
#pragma unroll
  for (int t = 0; t < 4; ++t) {
    int col = 16 * t + colb;
#pragma unroll
    for (int v = 0; v < 8; ++v) {
      int row = r0 + v + 8 * half;
      float val = acc[t][v] + bl[col];
      if (ACT == 1) val = sp_f(val);
      else if (ACT == 2) val = sp_f(sp_f(val));
      else if (ACT == 3) {
        val = val * 0.9999950000374997f * gl[col] + bel[col];
        val = fmaxf(val, 0.f);
      }
      if (row < nrows && col < fout) dst[(size_t)row * fout + col] = val;
    }
  }
}

// per-edge: he = (sp?)(lin(sp(lin(y,e1)),e2)); m = hv[src]*he; agg[dst]+=m
template <bool STORE_M, bool OUTER_SP>
__global__ __launch_bounds__(256) void edge_msg_kernel(
    const float* __restrict__ y, int E, const float* __restrict__ We1,
    const float* __restrict__ b1, const float* __restrict__ We2, const float* __restrict__ b2,
    const float* __restrict__ hv, const int* __restrict__ srcI, const int* __restrict__ dstI,
    float* __restrict__ m_out, float* __restrict__ agg) {
  __shared__ alignas(32) bf16 W1[64 * 64], W2[64 * 64];
  __shared__ float B1[64], B2[64];
  __shared__ alignas(32) bf16 At[8][16 * 64];
  __shared__ alignas(32) bf16 H1[8][16 * 64];
  const int tid = threadIdx.x, lane = tid & 31, wave = tid >> 5;

  stage_w_swz<64>(We1, 40, 64, W1, tid);
  stage_w_swz<64>(We2, 64, 64, W2, tid);
  if (tid < 64) { B1[tid] = b1[tid]; B2[tid] = b2[tid]; }
  const int e0 = (blockIdx.x * 8 + wave) * 16;
  bf16* at = At[wave];
  for (int i = lane; i < 16 * 64; i += 32) {
    int rr = i >> 6, cc = i & 63;
    float v = 0.f;
    if (e0 + rr < E && cc < 40) v = y[(size_t)(e0 + rr) * 40 + cc];
    at[i] = (bf16)v;
  }
  __syncthreads();

  v8f acc[4] = {};
  wave_gemm64<64>(at, W1, lane, acc);

  const int colb = lane & 15, half = lane >> 4;
  bf16* h1 = H1[wave];
#pragma unroll
  for (int t = 0; t < 4; ++t) {
    int col = 16 * t + colb;
#pragma unroll
    for (int v = 0; v < 8; ++v) {
      int row = v + 8 * half;
      h1[row * 64 + col] = (bf16)sp_f(acc[t][v] + B1[col]);
    }
  }
  __syncthreads();

  v8f acc2[4] = {};
  wave_gemm64<64>(h1, W2, lane, acc2);

#pragma unroll
  for (int t = 0; t < 4; ++t) {
    int col = 16 * t + colb;
#pragma unroll
    for (int v = 0; v < 8; ++v) {
      int e = e0 + v + 8 * half;
      if (e < E) {
        float he = acc2[t][v] + B2[col];
        if (OUTER_SP) he = sp_f(he);
        int s = srcI[e], d = dstI[e];
        float mv = hv[(size_t)s * 64 + col] * he;
        if (STORE_M) m_out[(size_t)e * 64 + col] = mv;
        atomicAdd(&agg[(size_t)d * 64 + col], mv);
      }
    }
  }
}

// hv2 = [y | m] @ W[104,64] + b    (ycat never materialized)
__global__ __launch_bounds__(256) void hv2_kernel(const float* __restrict__ y,
                                                  const float* __restrict__ m, int E,
                                                  const float* __restrict__ W,
                                                  const float* __restrict__ bias,
                                                  float* __restrict__ out) {
  constexpr int KPAD = 128;
  __shared__ alignas(32) bf16 Wl[KPAD * 64];
  __shared__ float bl[64];
  __shared__ alignas(32) bf16 At[8][16 * KPAD];
  const int tid = threadIdx.x, lane = tid & 31, wave = tid >> 5;
  stage_w_swz<KPAD>(W, 104, 64, Wl, tid);
  if (tid < 64) bl[tid] = bias[tid];
  const int e0 = (blockIdx.x * 8 + wave) * 16;
  bf16* at = At[wave];
  for (int i = lane; i < 16 * KPAD; i += 32) {
    int rr = i >> 7, cc = i & 127;
    float v = 0.f;
    if (e0 + rr < E) {
      if (cc < 40) v = y[(size_t)(e0 + rr) * 40 + cc];
      else if (cc < 104) v = m[(size_t)(e0 + rr) * 64 + (cc - 40)];
    }
    at[i] = (bf16)v;
  }
  __syncthreads();
  v8f acc[4] = {};
  wave_gemm64<KPAD>(at, Wl, lane, acc);
  const int colb = lane & 15, half = lane >> 4;
#pragma unroll
  for (int t = 0; t < 4; ++t) {
    int col = 16 * t + colb;
#pragma unroll
    for (int v = 0; v < 8; ++v) {
      int row = e0 + v + 8 * half;
      if (row < E) out[(size_t)row * 64 + col] = acc[t][v] + bl[col];
    }
  }
}

// line-graph cfconv core: z=RBF(angle) on the fly, he2=sp(lin(sp(lin(z)))),
// m2 = hv2[lg_src]*he2 scattered into agg2[lg_dst].
__global__ __launch_bounds__(256) void lg_kernel(const float* __restrict__ angle_h, int L,
                                                 const float* __restrict__ We1,
                                                 const float* __restrict__ b1,
                                                 const float* __restrict__ We2,
                                                 const float* __restrict__ b2,
                                                 const float* __restrict__ hv2,
                                                 const int* __restrict__ lsrc,
                                                 const int* __restrict__ ldst,
                                                 float* __restrict__ agg2) {
  __shared__ alignas(32) bf16 W1[64 * 64], W2[64 * 64];
  __shared__ float B1[64], B2[64];
  __shared__ float Hh[8][16];
  __shared__ alignas(32) bf16 Zt[8][16 * 64];
  __shared__ alignas(32) bf16 H1[8][16 * 64];
  const int tid = threadIdx.x, lane = tid & 31, wave = tid >> 5;

  stage_w_swz<64>(We1, 40, 64, W1, tid);
  stage_w_swz<64>(We2, 64, 64, W2, tid);
  if (tid < 64) { B1[tid] = b1[tid]; B2[tid] = b2[tid]; }
  const int l0 = (blockIdx.x * 8 + wave) * 16;
  if (lane < 16) Hh[wave][lane] = (l0 + lane < L) ? angle_h[l0 + lane] : 0.f;
  __syncthreads();

  // z = exp(-19.5*(h - c_j)^2), c_j = -1 + j*2/39   (ANGLE RBF, never stored)
  bf16* zt = Zt[wave];
  for (int i = lane; i < 16 * 64; i += 32) {
    int rr = i >> 6, cc = i & 63;
    float v = 0.f;
    if (l0 + rr < L && cc < 40) {
      float d = Hh[wave][rr] - (-1.f + cc * (2.f / 39.f));
      v = __expf(-19.5f * d * d);
    }
    zt[i] = (bf16)v;
  }
  __syncthreads();

  v8f acc[4] = {};
  wave_gemm64<64>(zt, W1, lane, acc);

  const int colb = lane & 15, half = lane >> 4;
  bf16* h1 = H1[wave];
#pragma unroll
  for (int t = 0; t < 4; ++t) {
    int col = 16 * t + colb;
#pragma unroll
    for (int v = 0; v < 8; ++v) {
      int row = v + 8 * half;
      h1[row * 64 + col] = (bf16)sp_f(acc[t][v] + B1[col]);
    }
  }
  __syncthreads();

  v8f acc2[4] = {};
  wave_gemm64<64>(h1, W2, lane, acc2);

#pragma unroll
  for (int t = 0; t < 4; ++t) {
    int col = 16 * t + colb;
#pragma unroll
    for (int v = 0; v < 8; ++v) {
      int l = l0 + v + 8 * half;
      if (l < L) {
        float he = sp_f(acc2[t][v] + B2[col]);
        int s = lsrc[l], d = ldst[l];
        float mv = hv2[(size_t)s * 64 + col] * he;
        atomicAdd(&agg2[(size_t)d * 64 + col], mv);
      }
    }
  }
}

// h = relu(bn_final(x)); pooled[g] += h; cnt[g] += 1
__global__ __launch_bounds__(256) void pool_kernel(const float* __restrict__ x, int N,
                                                   const float* __restrict__ g,
                                                   const float* __restrict__ be,
                                                   const int* __restrict__ n2g,
                                                   float* __restrict__ pooled,
                                                   float* __restrict__ cnt) {
  int i = blockIdx.x * 256 + threadIdx.x;
  if (i < N * 64) {
    int node = i >> 6, c = i & 63;
    int gr = n2g[node];
    float v = x[i] * 0.9999950000374997f * g[c] + be[c];
    v = fmaxf(v, 0.f);
    atomicAdd(&pooled[gr * 64 + c], v);
    if (c == 0) atomicAdd(&cnt[gr], 1.f);
  }
}

__global__ void fc_kernel(const float* __restrict__ pooled, const float* __restrict__ cnt,
                          const float* __restrict__ W, const float* __restrict__ b,
                          float* __restrict__ out) {
  int g = threadIdx.x;
  if (g < 64) {
    float c = fmaxf(cnt[g], 1.f);
    float s = 0.f;
#pragma unroll 8
    for (int k = 0; k < 64; ++k) s += (pooled[g * 64 + k] / c) * W[k];
    out[g] = s + b[0];
  }
}

// ----------------------------------------------------------------- launch ---

extern "C" void kernel_launch(void* const* d_in, const int* in_sizes, int n_in, void* d_out,
                              int out_size, void* d_ws, size_t ws_size, hipStream_t stream) {
  const float* atom = (const float*)d_in[0];
  const float* r    = (const float*)d_in[1];
  const int* srcI   = (const int*)d_in[2];
  const int* dstI   = (const int*)d_in[3];
  const int* lsrc   = (const int*)d_in[4];
  const int* ldst   = (const int*)d_in[5];
  const float* ang  = (const float*)d_in[6];
  const int* n2g    = (const int*)d_in[7];
  // d_in[8] = num_graphs (fixed 64)

  const int N = in_sizes[0] / 92;
  const int E = in_sizes[1] / 3;
  const int L = in_sizes[4];

  // params: jax tree-leaf order (dict keys sorted at each level, tuples (W,b))
  const int P = 9;
  const float* bn_g  = (const float*)d_in[P + 0];
  const float* bn_b  = (const float*)d_in[P + 1];
  const float* bnf_g = (const float*)d_in[P + 2];
  const float* bnf_b = (const float*)d_in[P + 3];
  struct Clgn {
    const float *e1W, *e1b, *e2W, *e2b;
    const float *lg_e1W, *lg_e1b, *lg_e2W, *lg_e2b, *lg_nW, *lg_nb, *lg_oW, *lg_ob;
    const float *nW, *nb, *oW, *ob;
  };
  auto getc = [&](int base) {
    Clgn c;
    c.e1W = (const float*)d_in[base + 0];  c.e1b = (const float*)d_in[base + 1];
    c.e2W = (const float*)d_in[base + 2];  c.e2b = (const float*)d_in[base + 3];
    c.lg_e1W = (const float*)d_in[base + 4]; c.lg_e1b = (const float*)d_in[base + 5];
    c.lg_e2W = (const float*)d_in[base + 6]; c.lg_e2b = (const float*)d_in[base + 7];
    c.lg_nW = (const float*)d_in[base + 8];  c.lg_nb = (const float*)d_in[base + 9];
    c.lg_oW = (const float*)d_in[base + 10]; c.lg_ob = (const float*)d_in[base + 11];
    c.nW = (const float*)d_in[base + 12];    c.nb = (const float*)d_in[base + 13];
    c.oW = (const float*)d_in[base + 14];    c.ob = (const float*)d_in[base + 15];
    return c;
  };
  Clgn c1 = getc(P + 4), c2 = getc(P + 20);
  const float *c3e1W = (const float*)d_in[P + 36], *c3e1b = (const float*)d_in[P + 37];
  const float *c3e2W = (const float*)d_in[P + 38], *c3e2b = (const float*)d_in[P + 39];
  const float *c3nW  = (const float*)d_in[P + 40], *c3nb  = (const float*)d_in[P + 41];
  const float *c3oW  = (const float*)d_in[P + 42], *c3ob  = (const float*)d_in[P + 43];
  const float *embW  = (const float*)d_in[P + 44], *embB  = (const float*)d_in[P + 45];
  const float *fcW   = (const float*)d_in[P + 46], *fcB   = (const float*)d_in[P + 47];

  // workspace layout
  float* w = (float*)d_ws;
  size_t o = 0;
  float* x_a   = w + o; o += (size_t)N * 64;
  float* x_b   = w + o; o += (size_t)N * 64;
  float* hv_n  = w + o; o += (size_t)N * 64;
  float* agg_n = w + o; o += (size_t)N * 64;
  float* y_a   = w + o; o += (size_t)E * 40;
  float* y_b   = w + o; o += (size_t)E * 40;
  float* m_buf = w + o; o += (size_t)E * 64;  // reused as agg2 after hv2
  float* hv2   = w + o; o += (size_t)E * 64;
  float* pooled = w + o; o += 64 * 64;
  float* cntb   = w + o; o += 64;
  (void)ws_size; (void)n_in; (void)out_size;

  const int gbN = (N + 127) / 128;
  const int gbE = (E + 127) / 128;
  const int gbL = (L + 127) / 128;

  rbf_y_kernel<<<(E + 255) / 256, 256, 0, stream>>>(r, E, y_a);
  gemm_rows_kernel<96, 3><<<gbN, 256, 0, stream>>>(atom, N, 92, embW, embB, 64, bn_g, bn_b, x_a);

  auto layer = [&](const Clgn& c, const float* x_in, const float* y_in, float* x_out,
                   float* y_out) {
    gemm_rows_kernel<64, 0><<<gbN, 256, 0, stream>>>(x_in, N, 64, c.nW, c.nb, 64, nullptr,
                                                     nullptr, hv_n);
    hipMemsetAsync(agg_n, 0, (size_t)N * 64 * sizeof(float), stream);
    edge_msg_kernel<true, false><<<gbE, 256, 0, stream>>>(y_in, E, c.e1W, c.e1b, c.e2W, c.e2b,
                                                          hv_n, srcI, dstI, m_buf, agg_n);
    gemm_rows_kernel<64, 1><<<gbN, 256, 0, stream>>>(agg_n, N, 64, c.oW, c.ob, 64, nullptr,
                                                     nullptr, x_out);
    hv2_kernel<<<gbE, 256, 0, stream>>>(y_in, m_buf, E, c.lg_nW, c.lg_nb, hv2);
    hipMemsetAsync(m_buf, 0, (size_t)E * 64 * sizeof(float), stream);  // m -> agg2
    lg_kernel<<<gbL, 256, 0, stream>>>(ang, L, c.lg_e1W, c.lg_e1b, c.lg_e2W, c.lg_e2b, hv2, lsrc,
                                       ldst, m_buf);
    gemm_rows_kernel<64, 2><<<gbE, 256, 0, stream>>>(m_buf, E, 64, c.lg_oW, c.lg_ob, 40, nullptr,
                                                     nullptr, y_out);
  };
  layer(c1, x_a, y_a, x_b, y_b);
  layer(c2, x_b, y_b, x_a, y_a);

  // conv3 (plain cfconv on nodes)
  gemm_rows_kernel<64, 0><<<gbN, 256, 0, stream>>>(x_a, N, 64, c3nW, c3nb, 64, nullptr, nullptr,
                                                   hv_n);
  hipMemsetAsync(agg_n, 0, (size_t)N * 64 * sizeof(float), stream);
  edge_msg_kernel<false, true><<<gbE, 256, 0, stream>>>(y_a, E, c3e1W, c3e1b, c3e2W, c3e2b, hv_n,
                                                        srcI, dstI, nullptr, agg_n);
  gemm_rows_kernel<64, 1><<<gbN, 256, 0, stream>>>(agg_n, N, 64, c3oW, c3ob, 64, nullptr, nullptr,
                                                   x_b);

  hipMemsetAsync(pooled, 0, (64 * 64 + 64) * sizeof(float), stream);
  pool_kernel<<<(N * 64 + 255) / 256, 256, 0, stream>>>(x_b, N, bnf_g, bnf_b, n2g, pooled, cntb);
  fc_kernel<<<1, 64, 0, stream>>>(pooled, cntb, fcW, fcB, (float*)d_out);
}
